// Attention_38027640439015
// MI455X (gfx1250) — matrix operand order repaired and checked by reading, not compiled, not run on hardware
//
#include <hip/hip_runtime.h>
#include <hip/hip_bf16.h>

typedef __attribute__((ext_vector_type(16))) __bf16 v16bf;
typedef __attribute__((ext_vector_type(8)))  __bf16 v8bf;
typedef __attribute__((ext_vector_type(8)))  float  v8f;

#define WMMA_BF16(a, b, c) \
  __builtin_amdgcn_wmma_f32_16x16x32_bf16(false, (a), false, (b), (short)0, (c), false, false)

#define CAT16(x0, x1) __builtin_shufflevector((x0), (x1), 0,1,2,3,4,5,6,7,8,9,10,11,12,13,14,15)

static constexpr int kB   = 2;
static constexpr int kT   = 2048;
static constexpr int kH   = 1024;
static constexpr int kNH  = 16;
static constexpr int kHD  = 64;
static constexpr int kM   = kB * kT;     // 4096 token rows
static constexpr int k3H  = 3 * kH;      // 3072
static constexpr float kNegInf = -3.0e38f;

// A-operand loader: 16x32 bf16 tile, lane holds row (ln), 16 contiguous
// elements at K = 2*koff.. (contiguous order, matches B operands).
__device__ __forceinline__ v16bf load_a16(const __bf16* __restrict__ row, int k, int koff) {
  v8bf x0 = *(const v8bf*)(row + k + 2 * koff);
  v8bf x1 = *(const v8bf*)(row + k + 2 * koff + 8);
  return CAT16(x0, x1);
}

// ---------------------------------------------------------------------------
// Shared 32x64 GEMM core over K=kH, ping-pong double buffered (no register
// rotation copies in the steady state).
// ---------------------------------------------------------------------------
__device__ __forceinline__ void gemm_32x64(
    const __bf16* __restrict__ a0p, const __bf16* __restrict__ a1p,
    const __bf16* const* __restrict__ w, int koff, v8f acc[2][4]) {
  v16bf A0[2], B0[4], A1[2], B1[4];
  A0[0] = load_a16(a0p, 0, koff);
  A0[1] = load_a16(a1p, 0, koff);
#pragma unroll
  for (int nt = 0; nt < 4; ++nt) B0[nt] = *(const v16bf*)(w[nt]);

#pragma unroll 1
  for (int k = 0; k < kH - 64; k += 64) {
    // load set1 @ k+32
    A1[0] = load_a16(a0p, k + 32, koff);
    A1[1] = load_a16(a1p, k + 32, koff);
#pragma unroll
    for (int nt = 0; nt < 4; ++nt) B1[nt] = *(const v16bf*)(w[nt] + k + 32);
    // consume set0 @ k
#pragma unroll
    for (int nt = 0; nt < 4; ++nt) {
      acc[0][nt] = WMMA_BF16(A0[0], B0[nt], acc[0][nt]);
      acc[1][nt] = WMMA_BF16(A0[1], B0[nt], acc[1][nt]);
    }
    // load set0 @ k+64
    A0[0] = load_a16(a0p, k + 64, koff);
    A0[1] = load_a16(a1p, k + 64, koff);
#pragma unroll
    for (int nt = 0; nt < 4; ++nt) B0[nt] = *(const v16bf*)(w[nt] + k + 64);
    // consume set1 @ k+32
#pragma unroll
    for (int nt = 0; nt < 4; ++nt) {
      acc[0][nt] = WMMA_BF16(A1[0], B1[nt], acc[0][nt]);
      acc[1][nt] = WMMA_BF16(A1[1], B1[nt], acc[1][nt]);
    }
  }
  // epilogue: set0 holds kH-64; load set1 @ kH-32 and finish both
  A1[0] = load_a16(a0p, kH - 32, koff);
  A1[1] = load_a16(a1p, kH - 32, koff);
#pragma unroll
  for (int nt = 0; nt < 4; ++nt) B1[nt] = *(const v16bf*)(w[nt] + kH - 32);
#pragma unroll
  for (int nt = 0; nt < 4; ++nt) {
    acc[0][nt] = WMMA_BF16(A0[0], B0[nt], acc[0][nt]);
    acc[1][nt] = WMMA_BF16(A0[1], B0[nt], acc[1][nt]);
  }
#pragma unroll
  for (int nt = 0; nt < 4; ++nt) {
    acc[0][nt] = WMMA_BF16(A1[0], B1[nt], acc[0][nt]);
    acc[1][nt] = WMMA_BF16(A1[1], B1[nt], acc[1][nt]);
  }
}

// ---------------------------------------------------------------------------
// Conversion kernels: fp32 -> bf16 (plain, and transposed for B operands)
// ---------------------------------------------------------------------------
__global__ void k_cvt_bf16(const float* __restrict__ src, __bf16* __restrict__ dst, int n) {
  int i = blockIdx.x * blockDim.x + threadIdx.x;
  if (i < n) dst[i] = (__bf16)src[i];
}

__global__ void k_transpose_bf16(const float* __restrict__ src, __bf16* __restrict__ dst,
                                 int rows, int cols) {
  // src[rows][cols] f32 -> dst[cols][rows] bf16
  int i = blockIdx.x * blockDim.x + threadIdx.x;
  if (i < rows * cols) {
    int r = i / cols, c = i % cols;
    dst[(size_t)c * rows + r] = (__bf16)src[i];
  }
}

// ---------------------------------------------------------------------------
// QKV GEMM (M=4096, N=3072, K=1024) + bias + RoPE epilogue.
// One wave -> 32x64 tile; column section of 64 == one head-section (q/k/v).
// ---------------------------------------------------------------------------
__global__ __launch_bounds__(256) void k_qkv_rope(
    const __bf16* __restrict__ X, const __bf16* __restrict__ WT,
    const float* __restrict__ bias,
    __bf16* __restrict__ Qb,   // [B][NH][T][64]
    __bf16* __restrict__ Kb,   // [B][NH][T][64]
    __bf16* __restrict__ Vt)   // [B][NH][64][T]
{
  const int lane = threadIdx.x & 31;
  const int wave = (blockIdx.x * 256 + threadIdx.x) >> 5;
  const int tileM = wave & 127;          // 128 row tiles of 32
  const int sec   = wave >> 7;           // 48 column sections of 64
  if (sec >= 48) return;
  const int m0 = tileM * 32;
  const int c0 = sec * 64;
  const int ln = lane & 15, half = lane >> 4;
  const int koff = half * 8;
  const int bk   = half * 16;

  const __bf16* arow0 = X + (size_t)(m0 + ln) * kH;
  const __bf16* arow1 = X + (size_t)(m0 + 16 + ln) * kH;
  const __bf16* wrow[4];
#pragma unroll
  for (int nt = 0; nt < 4; ++nt) wrow[nt] = WT + (size_t)(c0 + nt * 16 + ln) * kH + bk;

  v8f acc[2][4] = {};
  gemm_32x64(arow0, arow1, wrow, koff, acc);

  // bias (per output column)
#pragma unroll
  for (int nt = 0; nt < 4; ++nt) {
    float bb = bias[c0 + nt * 16 + ln];
#pragma unroll
    for (int mt = 0; mt < 2; ++mt)
#pragma unroll
      for (int i = 0; i < 8; ++i) acc[mt][nt][i] += bb;
  }

  const int head = c0 / 192;
  const int r    = c0 % 192;   // 0 -> q, 64 -> k, 128 -> v
  if (r == 128) {
    // V: store transposed [B][NH][64][T]
#pragma unroll
    for (int mt = 0; mt < 2; ++mt)
#pragma unroll
      for (int nt = 0; nt < 4; ++nt)
#pragma unroll
        for (int i = 0; i < 8; ++i) {
          int m = m0 + mt * 16 + i + 8 * half;
          int bi = m >> 11, t = m & (kT - 1);
          int d = nt * 16 + ln;
          Vt[((size_t)(bi * kNH + head) * kHD + d) * kT + t] = (__bf16)acc[mt][nt][i];
        }
  } else {
    // Q or K: apply RoPE. (d, d+32) sit in acc tiles (nt, nt+2), same lane slot.
    __bf16* dst = (r == 0) ? Qb : Kb;
#pragma unroll
    for (int nt = 0; nt < 2; ++nt) {
      int d = nt * 16 + ln;                 // 0..31
      float inv = __powf(10000.0f, -(float)d * (1.0f / 32.0f));
#pragma unroll
      for (int mt = 0; mt < 2; ++mt)
#pragma unroll
        for (int i = 0; i < 8; ++i) {
          int m = m0 + mt * 16 + i + 8 * half;
          int bi = m >> 11, t = m & (kT - 1);
          float ang = (float)t * inv;
          float sn, cs;
          __sincosf(ang, &sn, &cs);
          float lo = acc[mt][nt][i], hi = acc[mt][nt + 2][i];
          float o_lo = lo * cs - hi * sn;
          float o_hi = hi * cs + lo * sn;
          size_t base = ((size_t)(bi * kNH + head) * kT + t) * kHD;
          dst[base + d]      = (__bf16)o_lo;
          dst[base + d + 32] = (__bf16)o_hi;
        }
    }
  }
}

// ---------------------------------------------------------------------------
// Flash attention: one wave per (b, head, 32-query tile). Online softmax.
// ---------------------------------------------------------------------------
__global__ __launch_bounds__(256) void k_attn(
    const __bf16* __restrict__ Qb, const __bf16* __restrict__ Kb,
    const __bf16* __restrict__ Vt, const int* __restrict__ amask,
    __bf16* __restrict__ AO)   // [B*T][H] bf16
{
  __shared__ alignas(32) __bf16 plds[8][32][32];
  const int lane = threadIdx.x & 31;
  const int w    = threadIdx.x >> 5;
  const int wave = (blockIdx.x * 256 + threadIdx.x) >> 5;
  const int qt = wave & 63;              // 64 q-tiles of 32
  const int h  = (wave >> 6) & 15;
  const int bi = wave >> 10;
  if (bi >= kB) return;
  const int q0 = qt * 32;
  const int ln = lane & 15, half = lane >> 4;
  const int koff = half * 8;

  // Q A-operands: 2 M-subtiles x 2 K-steps (head dims 0..31 / 32..63)
  const __bf16* Qp = Qb + ((size_t)(bi * kNH + h) * kT + q0) * kHD;
  v16bf qa[2][2];
#pragma unroll
  for (int mt = 0; mt < 2; ++mt)
#pragma unroll
    for (int s = 0; s < 2; ++s)
      qa[mt][s] = load_a16(Qp + (size_t)(mt * 16 + ln) * kHD, s * 32, koff);

  v8f acc[2][4] = {};
  float mrow[2][8], srow[2][8];
#pragma unroll
  for (int mt = 0; mt < 2; ++mt)
#pragma unroll
    for (int i = 0; i < 8; ++i) { mrow[mt][i] = kNegInf; srow[mt][i] = 0.0f; }

  const __bf16* Kp = Kb + (size_t)(bi * kNH + h) * kT * kHD;
  const __bf16* Vp = Vt + (size_t)(bi * kNH + h) * kHD * kT;
  const float scale = 0.125f;   // 1/sqrt(64)

#pragma unroll 1
  for (int k0 = 0; k0 < q0 + 32; k0 += 32) {      // k0 <= q0 always
    // prefetch next block of K and V toward the WGP (speculative)
    __builtin_prefetch(Kp + (size_t)(k0 + 32 + ln) * kHD, 0, 1);
    __builtin_prefetch(Kp + (size_t)(k0 + 48 + ln) * kHD, 0, 1);
#pragma unroll
    for (int nt = 0; nt < 4; ++nt)
      __builtin_prefetch(Vp + (size_t)(nt * 16 + ln) * kT + k0 + 32, 0, 1);

    // issue V-tile loads early: latency hides under S-WMMAs + softmax VALU
    v16bf bv[4];
#pragma unroll
    for (int nt = 0; nt < 4; ++nt)
      bv[nt] = *(const v16bf*)(Vp + (size_t)(nt * 16 + ln) * kT + k0 + half * 16);

    // S = Q @ K^T : 2 M-subtiles x 2 token tiles, K-dim 64 = 2 steps
    v8f s0[2] = {}, s1[2] = {};
#pragma unroll
    for (int s = 0; s < 2; ++s) {
      v16bf kb0 = *(const v16bf*)(Kp + (size_t)(k0 + ln) * kHD + s * 32 + half * 16);
      v16bf kb1 = *(const v16bf*)(Kp + (size_t)(k0 + 16 + ln) * kHD + s * 32 + half * 16);
#pragma unroll
      for (int mt = 0; mt < 2; ++mt) {
        s0[mt] = WMMA_BF16(qa[mt][s], kb0, s0[mt]);
        s1[mt] = WMMA_BF16(qa[mt][s], kb1, s1[mt]);
      }
    }
    const int kt0 = k0 + ln, kt1 = k0 + 16 + ln;
    const bool ok0 = amask[bi * kT + kt0] != 0;
    const bool ok1 = amask[bi * kT + kt1] != 0;

    float corr[2][8];
#pragma unroll
    for (int mt = 0; mt < 2; ++mt)
#pragma unroll
      for (int i = 0; i < 8; ++i) {
        int qrow = q0 + mt * 16 + i + 8 * half;
        float v0 = (ok0 && kt0 <= qrow) ? s0[mt][i] * scale : kNegInf;
        float v1 = (ok1 && kt1 <= qrow) ? s1[mt][i] * scale : kNegInf;
        // row max over the 16 lanes of this half (rows never cross halves)
        float rm = fmaxf(v0, v1);
        rm = fmaxf(rm, __shfl_xor(rm, 1));
        rm = fmaxf(rm, __shfl_xor(rm, 2));
        rm = fmaxf(rm, __shfl_xor(rm, 4));
        rm = fmaxf(rm, __shfl_xor(rm, 8));
        float nm = fmaxf(mrow[mt][i], rm);
        float c  = __expf(mrow[mt][i] - nm);
        float p0 = __expf(v0 - nm);
        float p1 = __expf(v1 - nm);
        float rs = p0 + p1;
        rs += __shfl_xor(rs, 1);
        rs += __shfl_xor(rs, 2);
        rs += __shfl_xor(rs, 4);
        rs += __shfl_xor(rs, 8);
        srow[mt][i] = srow[mt][i] * c + rs;
        mrow[mt][i] = nm;
        corr[mt][i] = c;
        int prow = mt * 16 + i + 8 * half;
        plds[w][prow][ln]      = (__bf16)p0;
        plds[w][prow][16 + ln] = (__bf16)p1;
      }
#pragma unroll
    for (int mt = 0; mt < 2; ++mt)
#pragma unroll
      for (int nt = 0; nt < 4; ++nt)
#pragma unroll
        for (int i = 0; i < 8; ++i) acc[mt][nt][i] *= corr[mt][i];

    // Re-stripe P (C layout -> A operand layout) via LDS, wave-local
    // (DS ops are in-order within a wave; no block barrier needed).
    v16bf pa[2];
#pragma unroll
    for (int mt = 0; mt < 2; ++mt) {
      v8bf pa0 = *(const v8bf*)(&plds[w][mt * 16 + ln][2 * koff]);
      v8bf pa1 = *(const v8bf*)(&plds[w][mt * 16 + ln][2 * koff + 8]);
      pa[mt] = CAT16(pa0, pa1);
    }
    // acc += P @ V  (V transposed: contiguous along tokens)
#pragma unroll
    for (int nt = 0; nt < 4; ++nt)
#pragma unroll
      for (int mt = 0; mt < 2; ++mt)
        acc[mt][nt] = WMMA_BF16(pa[mt], bv[nt], acc[mt][nt]);
  }

  // Normalize and write attn output (bf16, token-major [B*T][H])
#pragma unroll
  for (int mt = 0; mt < 2; ++mt)
#pragma unroll
    for (int nt = 0; nt < 4; ++nt)
#pragma unroll
      for (int i = 0; i < 8; ++i) {
        int t = q0 + mt * 16 + i + 8 * half;
        size_t g = (size_t)bi * kT + t;
        AO[g * kH + h * kHD + nt * 16 + ln] = (__bf16)(acc[mt][nt][i] / srow[mt][i]);
      }
}

// ---------------------------------------------------------------------------
// Output projection (M=4096, N=1024, K=1024) -> f32, 32x64 per wave
// ---------------------------------------------------------------------------
__global__ __launch_bounds__(256) void k_outproj(
    const __bf16* __restrict__ A, const __bf16* __restrict__ WT,
    float* __restrict__ out)
{
  const int lane = threadIdx.x & 31;
  const int wave = (blockIdx.x * 256 + threadIdx.x) >> 5;
  const int tileM = wave & 127;
  const int sec   = wave >> 7;           // 16 sections of 64 cols
  if (sec >= 16) return;
  const int m0 = tileM * 32, c0 = sec * 64;
  const int ln = lane & 15, half = lane >> 4;
  const int koff = half * 8, bk = half * 16;

  const __bf16* arow0 = A + (size_t)(m0 + ln) * kH;
  const __bf16* arow1 = A + (size_t)(m0 + 16 + ln) * kH;
  const __bf16* wrow[4];
#pragma unroll
  for (int nt = 0; nt < 4; ++nt) wrow[nt] = WT + (size_t)(c0 + nt * 16 + ln) * kH + bk;

  v8f acc[2][4] = {};
  gemm_32x64(arow0, arow1, wrow, koff, acc);

#pragma unroll
  for (int mt = 0; mt < 2; ++mt)
#pragma unroll
    for (int nt = 0; nt < 4; ++nt)
#pragma unroll
      for (int i = 0; i < 8; ++i)
        out[(size_t)(m0 + mt * 16 + i + 8 * half) * kH + c0 + nt * 16 + ln] = acc[mt][nt][i];
}

// ---------------------------------------------------------------------------
extern "C" void kernel_launch(void* const* d_in, const int* in_sizes, int n_in,
                              void* d_out, int out_size, void* d_ws, size_t ws_size,
                              hipStream_t stream) {
  const float* hidden = (const float*)d_in[0];
  const int*   amask  = (const int*)d_in[1];
  const float* w_qkv  = (const float*)d_in[2];
  const float* b_qkv  = (const float*)d_in[3];
  const float* w_out  = (const float*)d_in[4];
  float* out = (float*)d_out;

  char* ws = (char*)d_ws;
  __bf16* Xbf   = (__bf16*)ws; ws += (size_t)kM * kH * 2;              // 8 MiB
  __bf16* WqkvT = (__bf16*)ws; ws += (size_t)k3H * kH * 2;             // 6 MiB
  __bf16* WoutT = (__bf16*)ws; ws += (size_t)kH * kH * 2;              // 2 MiB
  __bf16* Qb    = (__bf16*)ws; ws += (size_t)kB * kNH * kT * kHD * 2;  // 8 MiB
  __bf16* Kb    = (__bf16*)ws; ws += (size_t)kB * kNH * kT * kHD * 2;  // 8 MiB
  __bf16* Vt    = (__bf16*)ws; ws += (size_t)kB * kNH * kHD * kT * 2;  // 8 MiB
  __bf16* AO    = (__bf16*)ws; ws += (size_t)kM * kH * 2;              // 8 MiB

  // 1) fp32 -> bf16 conversions / transposes
  {
    int n = kM * kH;
    k_cvt_bf16<<<(n + 255) / 256, 256, 0, stream>>>(hidden, Xbf, n);
  }
  {
    int n = kH * k3H;
    k_transpose_bf16<<<(n + 255) / 256, 256, 0, stream>>>(w_qkv, WqkvT, kH, k3H);
  }
  {
    int n = kH * kH;
    k_transpose_bf16<<<(n + 255) / 256, 256, 0, stream>>>(w_out, WoutT, kH, kH);
  }

  // 2) QKV projection + bias + RoPE  (6144 waves -> 768 blocks of 8 waves)
  k_qkv_rope<<<768, 256, 0, stream>>>(Xbf, WqkvT, b_qkv, Qb, Kb, Vt);

  // 3) Attention (2048 waves -> 256 blocks)
  k_attn<<<256, 256, 0, stream>>>(Qb, Kb, Vt, amask, AO);

  // 4) Output projection (2048 waves -> 256 blocks)
  k_outproj<<<256, 256, 0, stream>>>(AO, WoutT, out);
}
